// NeuralMeshSimplification_64269890617539
// MI455X (gfx1250) — matrix operand order repaired and checked by reading, not compiled, not run on hardware
//
#include <hip/hip_runtime.h>
#include <hip/hip_bf16.h>
#include <math.h>

// Problem constants (from reference)
#define NND   10000              // nodes
#define MSA   5000               // sampled nodes (TARGET_RATIO * N)
#define KNN_K 15
#define NPAIR 105                // K*(K-1)/2
#define NTRI  (MSA * NPAIR)      // 525000 triangles
#define NEDGE 160000
#define DIMF  64

// d_out layout (flat floats, reference tuple order)
#define O_SIDX  0
#define O_SPRB  (MSA)
#define O_TRI   (2 * MSA)
#define O_TPRB  (O_TRI + 3 * NTRI)
#define O_FPRB  (O_TPRB + NTRI)
#define O_KEEP  (O_FPRB + NTRI)

typedef __attribute__((ext_vector_type(16))) _Float16 v16h;
typedef __attribute__((ext_vector_type(8)))  float    v8f;

__device__ __forceinline__ float nms_sigmoid(float z) { return 1.0f / (1.0f + expf(-z)); }

// ---------------------------------------------------------------------------
// Clear a float buffer (grid-stride)
__global__ void nms_clear_f32(float* __restrict__ p, int n) {
  for (int i = blockIdx.x * blockDim.x + threadIdx.x; i < n; i += gridDim.x * blockDim.x)
    p[i] = 0.0f;
}

// Zero small state (hist[256] + nv counter)
__global__ void nms_init_state(int* __restrict__ hist, int* __restrict__ nv) {
  int t = threadIdx.x;
  if (t < 256) hist[t] = 0;
  if (t == 0) *nv = 0;
}

// ---------------------------------------------------------------------------
// Fused dual GEMV via WMMA: D(16x16) = A(16x32 f16) x B(32x16 f16) + C, K=64.
// B col 0 = w_point, col 1 = w_face, cols 2..15 = 0 — staged once per block in
// LDS (col-major, 2 KB) so each lane reads one aligned v16h (ds_load_b128 x2).
// A rows loaded as unconditional float4 (global_load_b128); tail handled at
// wave granularity so EXEC is all-ones around the WMMA.
#define DOT_WAVES 4
__global__ void __launch_bounds__(32 * DOT_WAVES)
nms_dots_wmma(const float* __restrict__ x,
              const float* __restrict__ w_point,
              const float* __restrict__ w_face,
              float* __restrict__ point_prob,
              float* __restrict__ face_dot, int n) {
  __shared__ __align__(32) _Float16 Bs[16 * DIMF];   // [col][k], col-major

  const int tid = threadIdx.x;
  // zero all 16 columns
  for (int idx = tid; idx < 16 * DIMF; idx += 32 * DOT_WAVES)
    Bs[idx] = (_Float16)0.0f;
  __syncthreads();
  // columns 0 (w_point) and 1 (w_face): 128 values, one per thread
  if (tid < 2 * DIMF) {
    int k = tid & (DIMF - 1);
    int c = tid >> 6;
    float wv = c ? w_face[k] : w_point[k];
    Bs[c * DIMF + k] = (_Float16)wv;
  }
  __syncthreads();

  const int wave = tid >> 5;
  const int lane = tid & 31;
  const int tile = blockIdx.x * DOT_WAVES + wave;
  if (tile * 16 >= n) return;                        // wave-uniform tail skip

  const int half = lane >> 4;                        // 0: lanes 0-15, 1: lanes 16-31
  const int l15  = lane & 15;
  const int arow = tile * 16 + l15;                  // always < n (n % 16 == 0)
  const float4* xr4 = (const float4*)(x + (size_t)arow * DIMF);

  v8f acc = {};
  #pragma unroll
  for (int kc = 0; kc < 2; ++kc) {
    // A k order chosen here (contiguous, not the ISA table order): lane<16 holds K 0..15; lane>=16 K 16..31 (matches B)
    const int ka = kc * 32 + (half ? 16 : 0);        // 16B-aligned float runs
    float4 p0 = xr4[(ka >> 2) + 0];
    float4 p1 = xr4[(ka >> 2) + 1];
    float4 q0 = xr4[(ka >> 2) + 2];
    float4 q1 = xr4[(ka >> 2) + 3];
    v16h a;
    a[0]  = (_Float16)p0.x; a[1]  = (_Float16)p0.y; a[2]  = (_Float16)p0.z; a[3]  = (_Float16)p0.w;
    a[4]  = (_Float16)p1.x; a[5]  = (_Float16)p1.y; a[6]  = (_Float16)p1.z; a[7]  = (_Float16)p1.w;
    a[8]  = (_Float16)q0.x; a[9]  = (_Float16)q0.y; a[10] = (_Float16)q0.z; a[11] = (_Float16)q0.w;
    a[12] = (_Float16)q1.x; a[13] = (_Float16)q1.y; a[14] = (_Float16)q1.z; a[15] = (_Float16)q1.w;
    // B k order chosen here (contiguous, same as A): lanes 0-15 hold K=0..15 of column N=l15; lanes 16-31 K=16..31
    const int kb = kc * 32 + (half ? 16 : 0);        // 32B-aligned within column
    v16h b = *(const v16h*)(Bs + l15 * DIMF + kb);
    acc = __builtin_amdgcn_wmma_f32_16x16x32_f16(false, a, false, b, (short)0, acc,
                                                 false, false);
  }
  // D layout: lane holds rows M = half*8 + r at column N = l15. Need cols 0 and 1.
  if (l15 == 0) {
    #pragma unroll
    for (int r = 0; r < 8; ++r)
      point_prob[tile * 16 + half * 8 + r] = nms_sigmoid(acc[r]);
  } else if (l15 == 1) {
    #pragma unroll
    for (int r = 0; r < 8; ++r)
      face_dot[tile * 16 + half * 8 + r] = acc[r];
  }
}

// pos . w_pos per node (3-wide dot)
__global__ void nms_posdot(const float* __restrict__ pos, const float* __restrict__ w_pos,
                           float* __restrict__ pd, int n) {
  int i = blockIdx.x * blockDim.x + threadIdx.x;
  if (i < n)
    pd[i] = pos[3 * i] * w_pos[0] + pos[3 * i + 1] * w_pos[1] + pos[3 * i + 2] * w_pos[2];
}

// ---------------------------------------------------------------------------
// Scatter edges into dense adjacency. atomicMax on positive-float bits keeps
// duplicate-edge resolution deterministic (reference scatter order is unspecified).
__global__ void nms_scatter(const int* __restrict__ ei, const float* __restrict__ ep,
                            float* __restrict__ adj) {
  int e = blockIdx.x * blockDim.x + threadIdx.x;
  if (e < NEDGE) {
    int r = ei[e];
    int c = ei[NEDGE + e];
    atomicMax((int*)&adj[(size_t)r * MSA + c], __float_as_int(ep[e]));
  }
}

// ---------------------------------------------------------------------------
// Top-M of point_probs via rank-by-counting (stable: ties broken by lower index,
// matching jax.lax.top_k). LDS-tiled O(N^2) compares, N=10000.
__global__ void nms_rank(const float* __restrict__ probs, int* __restrict__ sidx,
                         float* __restrict__ out_idx, float* __restrict__ out_prob) {
  __shared__ float tile[256];
  int i = blockIdx.x * 256 + threadIdx.x;
  float p = (i < NND) ? probs[i] : 0.0f;
  int rank = 0;
  for (int base = 0; base < NND; base += 256) {
    int j = base + threadIdx.x;
    tile[threadIdx.x] = (j < NND) ? probs[j] : -3.0e38f;
    __syncthreads();
    int lim = min(256, NND - base);
    for (int t = 0; t < lim; ++t) {
      float q = tile[t];
      int jj = base + t;
      rank += ((q > p) || (q == p && jj < i)) ? 1 : 0;
    }
    __syncthreads();
  }
  if (i < NND && rank < MSA) {
    sidx[rank] = i;
    out_idx[rank]  = (float)i;
    out_prob[rank] = p;
  }
}

// g[m] = (x[sampled_idx[m]].w_face) + (pos[sampled_idx[m]].w_pos)
__global__ void nms_gather_g(const int* __restrict__ sidx, const float* __restrict__ fd,
                             const float* __restrict__ pd, float* __restrict__ g) {
  int m = blockIdx.x * blockDim.x + threadIdx.x;
  if (m < MSA) {
    int id = sidx[m];
    g[m] = fd[id] + pd[id];
  }
}

// ---------------------------------------------------------------------------
// Per-row top-15 of adjacency (stable insertion scan; ties keep lower column
// index, matching jax top_k). Rows are L2-resident (100 MB < 192 MB L2).
__global__ void nms_knn(const float* __restrict__ adj, int* __restrict__ knn) {
  int i = blockIdx.x * blockDim.x + threadIdx.x;
  if (i >= MSA) return;
  float v[KNN_K];
  int   ix[KNN_K];
  #pragma unroll
  for (int t = 0; t < KNN_K; ++t) { v[t] = -1.0f; ix[t] = 0; }
  const float* row = adj + (size_t)i * MSA;
  for (int c = 0; c < MSA; ++c) {
    if ((c & 255) == 0) __builtin_prefetch(row + c + 512, 0, 0);  // global_prefetch_b8
    float a = row[c];
    if (a > v[KNN_K - 1]) {
      int t = KNN_K - 1;
      while (t > 0 && v[t - 1] < a) { v[t] = v[t - 1]; ix[t] = ix[t - 1]; --t; }
      v[t] = a; ix[t] = c;
    }
  }
  #pragma unroll
  for (int t = 0; t < KNN_K; ++t) knn[i * KNN_K + t] = ix[t];
}

// ---------------------------------------------------------------------------
// Triangle formation: tri_probs, face_probs (via collapsed per-node dots), valid
// flags, and valid-count reduction.
__global__ void nms_tri(const float* __restrict__ adj, const int* __restrict__ knn,
                        const float* __restrict__ g, float* __restrict__ out_tri,
                        float* __restrict__ out_tp, float* __restrict__ out_fp,
                        unsigned char* __restrict__ valid, int* __restrict__ nv) {
  __shared__ int cnt;
  if (threadIdx.x == 0) cnt = 0;
  __syncthreads();
  int t = blockIdx.x * blockDim.x + threadIdx.x;
  int lv = 0;
  if (t < NTRI) {
    int i = t / NPAIR;
    int p = t % NPAIR;
    // decode (j,l) = p-th upper-triangular pair of K=15, row-major
    int j = 0, rem = p, span = KNN_K - 1;
    while (rem >= span) { rem -= span; --span; ++j; }
    int l = j + 1 + rem;
    int n1 = knn[i * KNN_K + j];
    int n2 = knn[i * KNN_K + l];
    float a1  = adj[(size_t)i  * MSA + n1];
    float a2  = adj[(size_t)i  * MSA + n2];
    float a12 = adj[(size_t)n1 * MSA + n2];
    bool val = a12 > 0.0f;
    float prod = a1 * a2 * a12;
    float tp = (val && prod > 0.0f) ? cbrtf(prod) : 0.0f;
    out_tri[3 * t + 0] = (float)i;
    out_tri[3 * t + 1] = (float)n1;
    out_tri[3 * t + 2] = (float)n2;
    out_tp[t] = tp;
    float fp = val ? nms_sigmoid((g[i] + g[n1] + g[n2]) * (1.0f / 3.0f)) : 0.0f;
    out_fp[t] = fp;
    valid[t] = val ? 1 : 0;
    lv = val ? 1 : 0;
  }
  if (lv) atomicAdd(&cnt, 1);
  __syncthreads();
  if (threadIdx.x == 0 && cnt) atomicAdd(nv, cnt);
}

// ---------------------------------------------------------------------------
// Masked-quantile (q = 0.5) via exact radix select on float bit patterns.
__global__ void nms_qsetup(const int* __restrict__ nv, unsigned* __restrict__ prefix,
                           int* __restrict__ kcur, float* __restrict__ frac) {
  int n = *nv;
  float pos = (n > 0) ? 0.5f * (float)(n - 1) : 0.0f;
  int klo = (int)floorf(pos);
  int khi = (int)ceilf(pos);
  if (n > 0) {
    if (klo > n - 1) klo = n - 1;
    if (khi > n - 1) khi = n - 1;
  }
  *frac = pos - floorf(pos);
  prefix[0] = 0u; prefix[1] = 0u;
  kcur[0] = klo;  kcur[1] = khi;
}

__global__ void nms_hist(const float* __restrict__ fp, const unsigned char* __restrict__ valid,
                         int* __restrict__ hist, const unsigned* __restrict__ prefixp, int pass) {
  __shared__ int h[256];
  h[threadIdx.x] = 0;
  __syncthreads();
  unsigned prefix = *prefixp;
  unsigned mask = (pass == 0) ? 0u : (0xFFFFFFFFu << (32 - 8 * pass));
  int shift = 24 - 8 * pass;
  for (int t = blockIdx.x * blockDim.x + threadIdx.x; t < NTRI; t += gridDim.x * blockDim.x) {
    if (valid[t]) {
      unsigned bits = __float_as_uint(fp[t]);  // sigmoid outputs >= 0: bit order == value order
      if ((bits & mask) == prefix) atomicAdd(&h[(bits >> shift) & 255], 1);
    }
  }
  __syncthreads();
  if (h[threadIdx.x]) atomicAdd(&hist[threadIdx.x], h[threadIdx.x]);
}

__global__ void nms_scan(int* __restrict__ hist, unsigned* __restrict__ prefix,
                         int* __restrict__ kcur, int pass, float* __restrict__ outval) {
  if (threadIdx.x == 0 && blockIdx.x == 0) {
    int k = *kcur;
    unsigned pre = *prefix;
    int shift = 24 - 8 * pass;
    int cum = 0, b = 255;
    for (int bb = 0; bb < 256; ++bb) {
      int c = hist[bb];
      if (cum + c > k) { b = bb; break; }
      cum += c;
    }
    *kcur = k - cum;
    pre |= ((unsigned)b) << shift;
    *prefix = pre;
    if (pass == 3 && outval) *outval = __uint_as_float(pre);
    for (int bb = 0; bb < 256; ++bb) hist[bb] = 0;  // ready for next pass
  }
}

__global__ void nms_thresh(const float* __restrict__ vlo, const float* __restrict__ vhi,
                           const float* __restrict__ frac, float* __restrict__ thr,
                           const int* __restrict__ nv) {
  if (threadIdx.x == 0 && blockIdx.x == 0) {
    if (*nv <= 0) { *thr = 3.0e38f; return; }
    float f = *frac;
    *thr = vlo[0] * (1.0f - f) + vhi[0] * f;
  }
}

__global__ void nms_keep(const float* __restrict__ fp, const unsigned char* __restrict__ valid,
                         const float* __restrict__ thr, float* __restrict__ out) {
  int t = blockIdx.x * blockDim.x + threadIdx.x;
  if (t < NTRI) out[t] = (valid[t] && fp[t] > *thr) ? 1.0f : 0.0f;
}

// ---------------------------------------------------------------------------
extern "C" void kernel_launch(void* const* d_in, const int* in_sizes, int n_in,
                              void* d_out, int out_size, void* d_ws, size_t ws_size,
                              hipStream_t stream) {
  (void)in_sizes; (void)n_in; (void)out_size; (void)ws_size;
  const float* x       = (const float*)d_in[0];
  const float* pos     = (const float*)d_in[1];
  const int*   ei      = (const int*)  d_in[2];
  const float* ep      = (const float*)d_in[3];
  const float* w_point = (const float*)d_in[4];
  const float* w_face  = (const float*)d_in[5];
  const float* w_pos   = (const float*)d_in[6];
  float* out = (float*)d_out;

  // Workspace carve-up (dominated by the 100 MB dense adjacency; L2-resident)
  char* w = (char*)d_ws;
  float* adj   = (float*)w; w += (size_t)MSA * MSA * sizeof(float);
  float* pprob = (float*)w; w += (size_t)NND * sizeof(float);
  float* fdot  = (float*)w; w += (size_t)NND * sizeof(float);
  float* pdot  = (float*)w; w += (size_t)NND * sizeof(float);
  float* g     = (float*)w; w += (size_t)MSA * sizeof(float);
  int*   sidx  = (int*)  w; w += (size_t)MSA * sizeof(int);
  int*   knn   = (int*)  w; w += (size_t)MSA * KNN_K * sizeof(int);
  unsigned char* valid = (unsigned char*)w; w += ((size_t)NTRI + 15) & ~(size_t)15;
  int*      nv     = (int*)     w; w += 16;
  int*      hist   = (int*)     w; w += 256 * sizeof(int);
  unsigned* prefix = (unsigned*)w; w += 2 * sizeof(unsigned);
  int*      kcur   = (int*)     w; w += 2 * sizeof(int);
  float*    selval = (float*)   w; w += 2 * sizeof(float);
  float*    frac   = (float*)   w; w += sizeof(float);
  float*    thr    = (float*)   w; w += sizeof(float);

  // 1) init
  nms_clear_f32<<<4096, 256, 0, stream>>>(adj, MSA * MSA);
  nms_init_state<<<1, 256, 0, stream>>>(hist, nv);

  // 2) fused dual GEMV (WMMA) + pos dot
  {
    int tiles = (NND + 15) / 16;                       // 625
    int blocks = (tiles + DOT_WAVES - 1) / DOT_WAVES;  // 157
    nms_dots_wmma<<<blocks, 32 * DOT_WAVES, 0, stream>>>(x, w_point, w_face, pprob, fdot, NND);
  }
  nms_posdot<<<(NND + 255) / 256, 256, 0, stream>>>(pos, w_pos, pdot, NND);

  // 3) adjacency scatter
  nms_scatter<<<(NEDGE + 255) / 256, 256, 0, stream>>>(ei, ep, adj);

  // 4) stable top-M point selection -> sampled_idx / sampled_probs
  nms_rank<<<(NND + 255) / 256, 256, 0, stream>>>(pprob, sidx, out + O_SIDX, out + O_SPRB);
  nms_gather_g<<<(MSA + 255) / 256, 256, 0, stream>>>(sidx, fdot, pdot, g);

  // 5) per-row top-15
  nms_knn<<<(MSA + 63) / 64, 64, 0, stream>>>(adj, knn);

  // 6) triangles + probabilities + valid count
  nms_tri<<<(NTRI + 255) / 256, 256, 0, stream>>>(adj, knn, g, out + O_TRI, out + O_TPRB,
                                                  out + O_FPRB, valid, nv);

  // 7) masked median threshold via dual radix select
  nms_qsetup<<<1, 1, 0, stream>>>(nv, prefix, kcur, frac);
  for (int sel = 0; sel < 2; ++sel) {
    for (int pass = 0; pass < 4; ++pass) {
      nms_hist<<<512, 256, 0, stream>>>(out + O_FPRB, valid, hist, prefix + sel, pass);
      nms_scan<<<1, 1, 0, stream>>>(hist, prefix + sel, kcur + sel, pass,
                                    (pass == 3) ? (selval + sel) : (float*)nullptr);
    }
  }
  nms_thresh<<<1, 1, 0, stream>>>(selval + 0, selval + 1, frac, thr, nv);

  // 8) keep mask
  nms_keep<<<(NTRI + 255) / 256, 256, 0, stream>>>(out + O_FPRB, valid, thr, out + O_KEEP);
}
